// GD_Unroll_62242666053730
// MI455X (gfx1250) — matrix operand order repaired and checked by reading, not compiled, not run on hardware
//
#include <hip/hip_runtime.h>
#include <hip/hip_bf16.h>

#define D 128

typedef __bf16        v16bf __attribute__((ext_vector_type(16)));
typedef float         v8f   __attribute__((ext_vector_type(8)));
typedef unsigned int  u32x8 __attribute__((ext_vector_type(8)));

// ---------- bf16 helpers (round-to-nearest-even) ----------
__device__ __forceinline__ unsigned short f2bf(float f) {
    unsigned int u = __float_as_uint(f);
    u += 0x7FFFu + ((u >> 16) & 1u);
    return (unsigned short)(u >> 16);
}
__device__ __forceinline__ unsigned int pkbf(float lo, float hi) {
    return (unsigned int)f2bf(lo) | ((unsigned int)f2bf(hi) << 16);
}

// ---------------------------------------------------------------------------
// Pack a 128x128 fp32 weight (row-major, W[k][n]) into the WMMA bf16 B-frag
// layout: out[((ntile*4 + c)*32 + lane)*8 + v] holds bf16 pair
//   K = 32c + 8g + 2(v&3) + 16(v>>2) (+0/+1),  col = ntile*16 + (lane&15),  g = lane>>4
// 8192 u32 per matrix; 5 matrices (W0, W1, Wq, Wk, Wv).
// ---------------------------------------------------------------------------
__global__ __launch_bounds__(256)
void pack_weights_kernel(const float* __restrict__ W0, const float* __restrict__ W1,
                         const float* __restrict__ Wq, const float* __restrict__ Wk,
                         const float* __restrict__ Wv, unsigned int* __restrict__ out)
{
    int tid = blockIdx.x * blockDim.x + threadIdx.x;
    if (tid >= 5 * 8192) return;
    int m     = tid >> 13;
    int t     = tid & 8191;
    int v     = t & 7;
    int lane  = (t >> 3) & 31;
    int c     = (t >> 8) & 3;
    int ntile = t >> 10;
    int col   = ntile * 16 + (lane & 15);
    int g     = lane >> 4;
    int k0    = 32 * c + 8 * g + 2 * (v & 3) + 16 * (v >> 2);
    const float* W = (m == 0) ? W0 : (m == 1) ? W1 : (m == 2) ? Wq : (m == 3) ? Wk : Wv;
    out[tid] = pkbf(W[k0 * D + col], W[(k0 + 1) * D + col]);
}

// ---------------------------------------------------------------------------
// agg[dst] += h[src] : one wave32 per edge, 4 floats per lane.
// ---------------------------------------------------------------------------
__global__ __launch_bounds__(256)
void scatter_agg_kernel(const float* __restrict__ h, const int* __restrict__ src,
                        const int* __restrict__ dst, float* __restrict__ agg, int ne)
{
    int wave = (int)((blockIdx.x * blockDim.x + threadIdx.x) >> 5);
    int lane = threadIdx.x & 31;
    if (wave >= ne) return;
    if (wave + 2048 < ne) {
        __builtin_prefetch(src + wave + 2048, 0, 1);
        __builtin_prefetch(dst + wave + 2048, 0, 1);
    }
    int s = src[wave], d = dst[wave];
    const float4 v = *(const float4*)(h + (size_t)s * D + lane * 4);
    float* a = agg + (size_t)d * D + lane * 4;
    atomicAdd(a + 0, v.x);
    atomicAdd(a + 1, v.y);
    atomicAdd(a + 2, v.z);
    atomicAdd(a + 3, v.w);
}

// ---------------------------------------------------------------------------
// Edge attention: alpha = <Q[dst], K[src]>;  out[dst] -= alpha * V[src]
// (x2 folded into x1 that already lives in `out`). One wave32 per edge.
// ---------------------------------------------------------------------------
__global__ __launch_bounds__(256)
void attn_edges_kernel(const float* __restrict__ Q, const float* __restrict__ Kb,
                       const float* __restrict__ V, const int* __restrict__ src,
                       const int* __restrict__ dst, float* __restrict__ out, int ne)
{
    int wave = (int)((blockIdx.x * blockDim.x + threadIdx.x) >> 5);
    int lane = threadIdx.x & 31;
    if (wave >= ne) return;
    if (wave + 2048 < ne) {
        __builtin_prefetch(src + wave + 2048, 0, 1);
        __builtin_prefetch(dst + wave + 2048, 0, 1);
    }
    int s = src[wave], d = dst[wave];
    const float4 q = *(const float4*)(Q  + (size_t)d * D + lane * 4);
    const float4 k = *(const float4*)(Kb + (size_t)s * D + lane * 4);
    float p = q.x * k.x + q.y * k.y + q.z * k.z + q.w * k.w;
    // wave32 all-reduce
    #pragma unroll
    for (int off = 16; off > 0; off >>= 1) p += __shfl_xor(p, off, 32);
    const float4 v = *(const float4*)(V + (size_t)s * D + lane * 4);
    float* o = out + (size_t)d * D + lane * 4;
    atomicAdd(o + 0, -p * v.x);
    atomicAdd(o + 1, -p * v.y);
    atomicAdd(o + 2, -p * v.z);
    atomicAdd(o + 3, -p * v.w);
}

// ---------- A-fragment load: 16x32 bf16 tile from a fp32 row-major matrix ---
// VGPR v in 0..3 holds K = 32c + 8g + {2v, 2v+1}; v in 4..7 adds +16.
// -> two contiguous 8-float segments at (32c + 8g) and (32c + 8g + 16).
__device__ __forceinline__ u32x8 load_afrag(const float* __restrict__ rowp,
                                            bool valid, int base)
{
    u32x8 u;
    if (valid) {
        const float4* q = (const float4*)(rowp + base);
        float4 a = q[0], b = q[1], c = q[4], d = q[5];
        u[0] = pkbf(a.x, a.y); u[1] = pkbf(a.z, a.w);
        u[2] = pkbf(b.x, b.y); u[3] = pkbf(b.z, b.w);
        u[4] = pkbf(c.x, c.y); u[5] = pkbf(c.z, c.w);
        u[6] = pkbf(d.x, d.y); u[7] = pkbf(d.z, d.w);
    } else {
        #pragma unroll
        for (int i = 0; i < 8; ++i) u[i] = 0u;
    }
    return u;
}

__device__ __forceinline__ v16bf load_bfrag(const uint4* __restrict__ pw,
                                            int ntile, int c, int lane)
{
    const uint4* p = pw + (size_t)(((ntile * 4 + c) * 32 + lane) * 2);
    uint4 a = p[0], b = p[1];
    u32x8 u;
    u[0] = a.x; u[1] = a.y; u[2] = a.z; u[3] = a.w;
    u[4] = b.x; u[5] = b.y; u[6] = b.z; u[7] = b.w;
    return __builtin_bit_cast(v16bf, u);
}

// ---------------------------------------------------------------------------
// Fused GEMM step:  x1 = h@W0 + agg@W1  (written to hnext),  Q/K/V = h@Wq/k/v
// One wave32 per 16-row tile, 4 waves (64 rows) per block.
// 8 N-tiles x 4 K-chunks x 5 WMMAs = 160 v_wmma_f32_16x16x32_bf16 per wave.
// ---------------------------------------------------------------------------
__global__ __launch_bounds__(128)
void gemm_step_kernel(const float* __restrict__ h, const float* __restrict__ agg,
                      const uint4* __restrict__ pW0, const uint4* __restrict__ pW1,
                      const uint4* __restrict__ pWq, const uint4* __restrict__ pWk,
                      const uint4* __restrict__ pWv,
                      float* __restrict__ x1, float* __restrict__ Q,
                      float* __restrict__ Kb, float* __restrict__ V, int n)
{
    const int wave = threadIdx.x >> 5;
    const int lane = threadIdx.x & 31;
    const int tile = blockIdx.x * 4 + wave;
    const int row_base = tile * 16;
    if (row_base >= n) return;          // wave-uniform: EXEC stays all-1 for WMMA
    const int r = lane & 15;
    const int g = lane >> 4;
    const int node = row_base + r;
    const bool valid = node < n;

    const float* hrow = h   + (size_t)node * D;
    const float* arow = agg + (size_t)node * D;

    u32x8 aH[4], aA[4];
    #pragma unroll
    for (int c = 0; c < 4; ++c) {
        int base = 32 * c + 8 * g;
        aH[c] = load_afrag(hrow, valid, base);
        aA[c] = load_afrag(arow, valid, base);
    }

    #pragma unroll
    for (int ntile = 0; ntile < 8; ++ntile) {
        v8f acc1 = {}; v8f accQ = {}; v8f accK = {}; v8f accV = {};
        #pragma unroll
        for (int c = 0; c < 4; ++c) {
            v16bf ah = __builtin_bit_cast(v16bf, aH[c]);
            v16bf aa = __builtin_bit_cast(v16bf, aA[c]);
            v16bf b0 = load_bfrag(pW0, ntile, c, lane);
            acc1 = __builtin_amdgcn_wmma_f32_16x16x32_bf16(false, ah, false, b0,
                                                           (short)0, acc1, false, false);
            v16bf b1 = load_bfrag(pW1, ntile, c, lane);
            acc1 = __builtin_amdgcn_wmma_f32_16x16x32_bf16(false, aa, false, b1,
                                                           (short)0, acc1, false, false);
            v16bf bq = load_bfrag(pWq, ntile, c, lane);
            accQ = __builtin_amdgcn_wmma_f32_16x16x32_bf16(false, ah, false, bq,
                                                           (short)0, accQ, false, false);
            v16bf bk = load_bfrag(pWk, ntile, c, lane);
            accK = __builtin_amdgcn_wmma_f32_16x16x32_bf16(false, ah, false, bk,
                                                           (short)0, accK, false, false);
            v16bf bv = load_bfrag(pWv, ntile, c, lane);
            accV = __builtin_amdgcn_wmma_f32_16x16x32_bf16(false, ah, false, bv,
                                                           (short)0, accV, false, false);
        }
        // C/D layout: VGPR v -> row M = v + 8g, col N = lane&15
        int col = ntile * 16 + r;
        #pragma unroll
        for (int v = 0; v < 8; ++v) {
            int row = row_base + v + 8 * g;
            if (row < n) {
                size_t o = (size_t)row * D + col;
                x1[o] = acc1[v];
                Q[o]  = accQ[v];
                Kb[o] = accK[v];
                V[o]  = accV[v];
            }
        }
    }
}

// ---------------------------------------------------------------------------
extern "C" void kernel_launch(void* const* d_in, const int* in_sizes, int n_in,
                              void* d_out, int out_size, void* d_ws, size_t ws_size,
                              hipStream_t stream)
{
    const float* x   = (const float*)d_in[0];
    const int*   ei  = (const int*)d_in[1];   // [2, E1] flat: src = ei, dst = ei+E1
    const int*   ei2 = (const int*)d_in[2];
    const float* W0  = (const float*)d_in[3];
    const float* W1  = (const float*)d_in[4];
    const float* Wq  = (const float*)d_in[5];
    const float* Wk  = (const float*)d_in[6];
    const float* Wv  = (const float*)d_in[7];

    const int n  = in_sizes[0] / D;   // 10000
    const int e1 = in_sizes[1] / 2;   // 320000
    const int e2 = in_sizes[2] / 2;   // 320000
    float* out = (float*)d_out;

    char* ws = (char*)d_ws;
    const size_t buf = (size_t)n * D * sizeof(float);
    float*        agg = (float*)(ws + 0 * buf);
    float*        Qb  = (float*)(ws + 1 * buf);
    float*        Kb  = (float*)(ws + 2 * buf);
    float*        Vb  = (float*)(ws + 3 * buf);
    float*        hA  = (float*)(ws + 4 * buf);
    float*        hB  = (float*)(ws + 5 * buf);
    unsigned int* pw  = (unsigned int*)(ws + 6 * buf);   // 5 * 8192 u32 = 160 KB

    // one-time per call: pack weights into WMMA bf16 B-fragment order
    pack_weights_kernel<<<(5 * 8192 + 255) / 256, 256, 0, stream>>>(W0, W1, Wq, Wk, Wv, pw);
    const uint4* pW0 = (const uint4*)(pw + 0 * 8192);
    const uint4* pW1 = (const uint4*)(pw + 1 * 8192);
    const uint4* pWq = (const uint4*)(pw + 2 * 8192);
    const uint4* pWk = (const uint4*)(pw + 3 * 8192);
    const uint4* pWv = (const uint4*)(pw + 4 * 8192);

    const int tiles       = (n + 15) / 16;
    const int gemm_blocks = (tiles + 3) / 4;
    const int e1_blocks   = (e1 + 7) / 8;   // 8 wave32 per 256-thread block
    const int e2_blocks   = (e2 + 7) / 8;

    const float* hcur = x;
    float* pp[2] = { hA, hB };
    for (int s = 0; s < 5; ++s) {
        float* hnext = (s == 4) ? out : pp[s & 1];
        hipMemsetAsync(agg, 0, buf, stream);
        scatter_agg_kernel<<<e1_blocks, 256, 0, stream>>>(hcur, ei, ei + e1, agg, e1);
        gemm_step_kernel<<<gemm_blocks, 128, 0, stream>>>(hcur, agg, pW0, pW1, pWq, pWk, pWv,
                                                          hnext, Qb, Kb, Vb, n);
        attn_edges_kernel<<<e2_blocks, 256, 0, stream>>>(Qb, Kb, Vb, ei2, ei2 + e2, hnext, e2);
        hcur = hnext;
    }
}
